// MambaSSM_74852690034785
// MI455X (gfx1250) — 8-bit matrix operand order repaired and checked by reading, compiled, not run on hardware
//
#include <hip/hip_runtime.h>
#include <hip/hip_bf16.h>
#include <math.h>

typedef int v8i __attribute__((ext_vector_type(8)));

#define D_MODEL 1024
#define D_STATE 16
#define D_CONV  4
#define D_INNER 2048
#define BSZ     2
#define LSEQ    2048
#define MROWS   (BSZ * LSEQ)

static constexpr size_t SZ_WIN  = 256ull * 16 * 1024;
static constexpr size_t SZ_WOUT = 64ull  * 32 * 1024;
static constexpr size_t SZ_WSML = 3ull   * 32 * 1024;
static constexpr size_t SZ_SCAL = 2048;
static constexpr size_t SZ_A1   = 256ull * 16 * 1024;
static constexpr size_t SZ_IS1  = 16384;
static constexpr size_t SZ_XIN  = (size_t)MROWS * D_INNER * 4;
static constexpr size_t SZ_XCV  = (size_t)MROWS * D_INNER * 4;
static constexpr size_t SZ_A2   = 256ull * 32 * 1024;
static constexpr size_t SZ_IS2  = 16384;
static constexpr size_t SZ_SM   = (size_t)MROWS * D_STATE * 4;
static constexpr size_t SZ_SIG  = 16384;
static constexpr size_t SZ_YQ   = 256ull * 32 * 1024;

static constexpr size_t OFF_WIN  = 0;
static constexpr size_t OFF_WOUT = OFF_WIN  + SZ_WIN;
static constexpr size_t OFF_WSML = OFF_WOUT + SZ_WOUT;
static constexpr size_t OFF_SCAL = OFF_WSML + SZ_WSML;
static constexpr size_t OFF_A1   = OFF_SCAL + SZ_SCAL;
static constexpr size_t OFF_IS1  = OFF_A1   + SZ_A1;
static constexpr size_t OFF_XIN  = OFF_IS1  + SZ_IS1;
static constexpr size_t OFF_XCV  = OFF_XIN  + SZ_XIN;
static constexpr size_t OFF_A2   = OFF_XCV  + SZ_XCV;
static constexpr size_t OFF_IS2  = OFF_A2   + SZ_A2;
static constexpr size_t OFF_DLT  = OFF_IS2  + SZ_IS2;
static constexpr size_t OFF_BM   = OFF_DLT  + SZ_SM;
static constexpr size_t OFF_CM   = OFF_BM   + SZ_SM;
static constexpr size_t OFF_SIG  = OFF_CM   + SZ_SM;
static constexpr size_t OFF_YQ   = OFF_SIG  + SZ_SIG;
static constexpr size_t OFF_IS3  = OFF_YQ   + SZ_YQ;

#define WMMA_IU8(A, B, C) \
  __builtin_amdgcn_wmma_i32_16x16x64_iu8(true, (A), true, (B), (C), false, false)

__global__ __launch_bounds__(256) void k_abs_partial(const float* __restrict__ w, int n,
                                                     float* __restrict__ partial) {
  __shared__ float red[256];
  float s = 0.f;
  for (int i = blockIdx.x * 256 + threadIdx.x; i < n; i += gridDim.x * 256)
    s += fabsf(w[i]);
  red[threadIdx.x] = s; __syncthreads();
  for (int o = 128; o > 0; o >>= 1) {
    if ((int)threadIdx.x < o) red[threadIdx.x] += red[threadIdx.x + o];
    __syncthreads();
  }
  if (threadIdx.x == 0) partial[blockIdx.x] = red[0];
}

__global__ __launch_bounds__(256) void k_abs_final(const float* __restrict__ partial, int nb,
                                                   float inv_count, float* __restrict__ scale_out) {
  __shared__ float red[256];
  red[threadIdx.x] = ((int)threadIdx.x < nb) ? partial[threadIdx.x] : 0.f;
  __syncthreads();
  for (int o = 128; o > 0; o >>= 1) {
    if ((int)threadIdx.x < o) red[threadIdx.x] += red[threadIdx.x + o];
    __syncthreads();
  }
  if (threadIdx.x == 0) scale_out[0] = fmaxf(red[0] * inv_count, 1e-5f);
}

__global__ __launch_bounds__(256) void k_wquant(const float* __restrict__ w,
                                                const float* __restrict__ scale,
                                                int K, int Kt, signed char* __restrict__ frag) {
  int idx = blockIdx.x * 256 + threadIdx.x;
  int n = idx / K, k = idx % K;
  float sc = scale[0];
  float q = fminf(fmaxf(rintf(w[idx] / sc), -1.f), 1.f);
  int nt = n >> 4, n4 = n & 15, kt = k >> 6, k6 = k & 63;
  int half32 = k6 >> 4, r = k6 & 15, rr = r & 7;
  int h = r >> 3;
  int j = half32 * 2 + (rr >> 2), bb = rr & 3, lane = h * 16 + n4;
  frag[(size_t)(nt * Kt + kt) * 1024 + lane * 32 + j * 4 + bb] = (signed char)q;
}

__global__ __launch_bounds__(256) void k_aquant(const float* __restrict__ x, int K, int Kt,
                                                signed char* __restrict__ frag,
                                                float* __restrict__ inv_s) {
  __shared__ float red[256];
  int m = blockIdx.x;
  const float* row = x + (size_t)m * K;
  float mx = 0.f;
  for (int k = threadIdx.x; k < K; k += 256) mx = fmaxf(mx, fabsf(row[k]));
  red[threadIdx.x] = mx; __syncthreads();
  for (int o = 128; o > 0; o >>= 1) {
    if ((int)threadIdx.x < o) red[threadIdx.x] = fmaxf(red[threadIdx.x], red[threadIdx.x + o]);
    __syncthreads();
  }
  float amax = fmaxf(red[0], 1e-5f);
  float s = 127.f / amax;
  if (threadIdx.x == 0) inv_s[m] = amax / 127.f;
  int mt = m >> 4, m4 = m & 15;
  for (int k = threadIdx.x; k < K; k += 256) {
    float q = fminf(fmaxf(rintf(row[k] * s), -128.f), 127.f);
    int kt = k >> 6, k6 = k & 63;
    int g = k6 >> 4, r = k6 & 15, h = r >> 3, rr = r & 7, p = rr >> 2, bb = rr & 3;
    int j = g * 2 + p, lane = h * 16 + m4;
    frag[(size_t)(mt * Kt + kt) * 1024 + lane * 32 + j * 4 + bb] = (signed char)q;
  }
}

__global__ __launch_bounds__(256) void k_gemm_gated(const signed char* __restrict__ afrag,
                                                    const signed char* __restrict__ bfrag,
                                                    const float* __restrict__ inv_s,
                                                    const float* __restrict__ wscale,
                                                    float* __restrict__ xout) {
  const int Kt  = 16;
  const int CT  = D_INNER / 16;
  const int MT2 = MROWS / 32;
  const int CT2 = CT / 2;
  int wave = blockIdx.x * 8 + (threadIdx.x >> 5);
  int lane = threadIdx.x & 31;
  if (wave >= MT2 * CT2) return;
  int mt0 = (wave / CT2) * 2, ct0 = (wave % CT2) * 2;
  const v8i* a0  = (const v8i*)(afrag + (size_t)(mt0    ) * Kt * 1024) + lane;
  const v8i* a1  = (const v8i*)(afrag + (size_t)(mt0 + 1) * Kt * 1024) + lane;
  const v8i* bg0 = (const v8i*)(bfrag + (size_t)(ct0         ) * Kt * 1024) + lane;
  const v8i* bg1 = (const v8i*)(bfrag + (size_t)(ct0 + 1     ) * Kt * 1024) + lane;
  const v8i* bv0 = (const v8i*)(bfrag + (size_t)(ct0 + CT    ) * Kt * 1024) + lane;
  const v8i* bv1 = (const v8i*)(bfrag + (size_t)(ct0 + CT + 1) * Kt * 1024) + lane;
  v8i z = {0, 0, 0, 0, 0, 0, 0, 0};
  v8i cg00 = z, cg01 = z, cg10 = z, cg11 = z;
  v8i cv00 = z, cv01 = z, cv10 = z, cv11 = z;
  for (int kt = 0; kt < Kt; ++kt) {
    v8i A0 = a0[kt * 32];
    v8i A1 = a1[kt * 32];
    v8i G0 = bg0[kt * 32];
    v8i G1 = bg1[kt * 32];
    v8i V0 = bv0[kt * 32];
    v8i V1 = bv1[kt * 32];
    cg00 = WMMA_IU8(A0, G0, cg00);
    cg01 = WMMA_IU8(A0, G1, cg01);
    cg10 = WMMA_IU8(A1, G0, cg10);
    cg11 = WMMA_IU8(A1, G1, cg11);
    cv00 = WMMA_IU8(A0, V0, cv00);
    cv01 = WMMA_IU8(A0, V1, cv01);
    cv10 = WMMA_IU8(A1, V0, cv10);
    cv11 = WMMA_IU8(A1, V1, cv11);
  }
  int n4 = lane & 15;
  int hi8 = (lane >> 4) << 3;
  float ws = wscale[0];
#pragma unroll
  for (int mi = 0; mi < 2; ++mi) {
#pragma unroll
    for (int ci = 0; ci < 2; ++ci) {
      const v8i& cg = mi ? (ci ? cg11 : cg10) : (ci ? cg01 : cg00);
      const v8i& cv = mi ? (ci ? cv11 : cv10) : (ci ? cv01 : cv00);
      int c = (ct0 + ci) * 16 + n4;
#pragma unroll
      for (int r = 0; r < 8; ++r) {
        int m = (mt0 + mi) * 16 + hi8 + r;
        float sc = inv_s[m] * ws;
        float g = (float)cg[r] * sc;
        float v = (float)cv[r] * sc;
        xout[(size_t)m * D_INNER + c] = v * (g / (1.f + __expf(-g)));
      }
    }
  }
}

__global__ __launch_bounds__(256) void k_conv(const float* __restrict__ xin,
                                              const float* __restrict__ cw,
                                              const float* __restrict__ cb,
                                              float* __restrict__ xout) {
  size_t idx = (size_t)blockIdx.x * 256 + threadIdx.x;
  int c = (int)(idx % D_INNER);
  int l = (int)((idx / D_INNER) % LSEQ);
  int b = (int)(idx / ((size_t)D_INNER * LSEQ));
  float acc = cb[c];
#pragma unroll
  for (int i = 0; i < D_CONV; ++i) {
    int t = l - (D_CONV - 1) + i;
    if (t >= 0) acc += cw[c * D_CONV + i] * xin[(size_t)(b * LSEQ + t) * D_INNER + c];
  }
  xout[idx] = acc;
}

__global__ __launch_bounds__(256) void k_gemm_small(const signed char* __restrict__ afrag,
                                                    const signed char* __restrict__ bfrag,
                                                    const float* __restrict__ inv_s,
                                                    const float* __restrict__ wscales,
                                                    const float* __restrict__ delta_b,
                                                    float* __restrict__ dlt,
                                                    float* __restrict__ bm,
                                                    float* __restrict__ cm) {
  const int Kt = 32;
  int wave = blockIdx.x * 8 + (threadIdx.x >> 5);
  int lane = threadIdx.x & 31;
  if (wave >= (MROWS / 16) * 3) return;
  int mt = wave / 3, which = wave % 3;
  const v8i* ap = (const v8i*)(afrag + (size_t)mt * Kt * 1024) + lane;
  const v8i* bp = (const v8i*)(bfrag + (size_t)which * Kt * 1024) + lane;
  v8i acc = {0, 0, 0, 0, 0, 0, 0, 0};
  for (int kt = 0; kt < Kt; ++kt)
    acc = WMMA_IU8(ap[kt * 32], bp[kt * 32], acc);
  float ws = wscales[which];
  float* dst = (which == 0) ? dlt : ((which == 1) ? bm : cm);
  int n = lane & 15;
  int mbase = mt * 16 + ((lane >> 4) << 3);
#pragma unroll
  for (int r = 0; r < 8; ++r) {
    int m = mbase + r;
    float v = (float)acc[r] * inv_s[m] * ws;
    if (which == 0) v += delta_b[n];
    dst[m * D_STATE + n] = v;
  }
}

__global__ __launch_bounds__(256) void k_scan(const float* __restrict__ dlt,
                                              const float* __restrict__ bm,
                                              const float* __restrict__ cm,
                                              const float* __restrict__ logA,
                                              float* __restrict__ sig) {
  __shared__ float As[256][D_STATE];
  __shared__ float Bs[256][D_STATE];
  const int CH = LSEQ / 256;
  int b = blockIdx.x, tid = threadIdx.x;
  float lA[D_STATE];
#pragma unroll
  for (int n = 0; n < D_STATE; ++n) lA[n] = logA[n];

  float Ap[D_STATE], Bc[D_STATE];
#pragma unroll
  for (int n = 0; n < D_STATE; ++n) { Ap[n] = 1.f; Bc[n] = 0.f; }
  int base = b * LSEQ + tid * CH;
  for (int i = 0; i < CH; ++i) {
    int m = base + i;
#pragma unroll
    for (int n = 0; n < D_STATE; ++n) {
      float d = dlt[m * D_STATE + n];
      float a = __expf(d * lA[n]);
      Ap[n] *= a;
      Bc[n] = a * Bc[n] + d * bm[m * D_STATE + n];
    }
  }
#pragma unroll
  for (int n = 0; n < D_STATE; ++n) { As[tid][n] = Ap[n]; Bs[tid][n] = Bc[n]; }
  __syncthreads();

  for (int off = 1; off < 256; off <<= 1) {
    float tA[D_STATE], tB[D_STATE];
    if (tid >= off) {
#pragma unroll
      for (int n = 0; n < D_STATE; ++n) {
        tA[n] = Ap[n] * As[tid - off][n];
        tB[n] = Ap[n] * Bs[tid - off][n] + Bc[n];
      }
    }
    __syncthreads();
    if (tid >= off) {
#pragma unroll
      for (int n = 0; n < D_STATE; ++n) {
        Ap[n] = tA[n]; Bc[n] = tB[n];
        As[tid][n] = tA[n]; Bs[tid][n] = tB[n];
      }
    }
    __syncthreads();
  }

  float h[D_STATE];
#pragma unroll
  for (int n = 0; n < D_STATE; ++n) h[n] = tid ? Bs[tid - 1][n] : 0.f;
  for (int i = 0; i < CH; ++i) {
    int m = base + i;
    float s = 0.f;
#pragma unroll
    for (int n = 0; n < D_STATE; ++n) {
      float d = dlt[m * D_STATE + n];
      float a = __expf(d * lA[n]);
      h[n] = a * h[n] + d * bm[m * D_STATE + n];
      s += cm[m * D_STATE + n] * h[n];
    }
    sig[m] = s;
  }
}

__global__ __launch_bounds__(256) void k_yquant(const float* __restrict__ xcv,
                                                const float* __restrict__ sig,
                                                const float* __restrict__ Dvec,
                                                signed char* __restrict__ frag,
                                                float* __restrict__ inv_s) {
  __shared__ float buf[D_INNER];
  __shared__ float red[256];
  int m = blockIdx.x;
  float sg = 1.f / (1.f + __expf(-sig[m]));
  const float* row = xcv + (size_t)m * D_INNER;
  float mx = 0.f;
  for (int k = threadIdx.x; k < D_INNER; k += 256) {
    float y = row[k] * (sg + Dvec[k]);
    buf[k] = y;
    mx = fmaxf(mx, fabsf(y));
  }
  red[threadIdx.x] = mx; __syncthreads();
  for (int o = 128; o > 0; o >>= 1) {
    if ((int)threadIdx.x < o) red[threadIdx.x] = fmaxf(red[threadIdx.x], red[threadIdx.x + o]);
    __syncthreads();
  }
  float amax = fmaxf(red[0], 1e-5f);
  float s = 127.f / amax;
  if (threadIdx.x == 0) inv_s[m] = amax / 127.f;
  int mt = m >> 4, m4 = m & 15;
  const int Kt = 32;
  for (int k = threadIdx.x; k < D_INNER; k += 256) {
    float q = fminf(fmaxf(rintf(buf[k] * s), -128.f), 127.f);
    int kt = k >> 6, k6 = k & 63;
    int g = k6 >> 4, r = k6 & 15, h = r >> 3, rr = r & 7, p = rr >> 2, bb = rr & 3;
    int j = g * 2 + p, lane = h * 16 + m4;
    frag[(size_t)(mt * Kt + kt) * 1024 + lane * 32 + j * 4 + bb] = (signed char)q;
  }
}

__global__ __launch_bounds__(256) void k_gemm_out(const signed char* __restrict__ afrag,
                                                  const signed char* __restrict__ bfrag,
                                                  const float* __restrict__ inv_s,
                                                  const float* __restrict__ wscale,
                                                  float* __restrict__ out) {
  const int Kt  = 32;
  const int NT  = D_MODEL / 16;
  const int MT2 = MROWS / 32;
  const int NT4 = NT / 4;
  int wave = blockIdx.x * 8 + (threadIdx.x >> 5);
  int lane = threadIdx.x & 31;
  if (wave >= MT2 * NT4) return;
  int mt0 = (wave / NT4) * 2, nt0 = (wave % NT4) * 4;
  const v8i* a0 = (const v8i*)(afrag + (size_t)(mt0    ) * Kt * 1024) + lane;
  const v8i* a1 = (const v8i*)(afrag + (size_t)(mt0 + 1) * Kt * 1024) + lane;
  const v8i* b0 = (const v8i*)(bfrag + (size_t)(nt0    ) * Kt * 1024) + lane;
  const v8i* b1 = (const v8i*)(bfrag + (size_t)(nt0 + 1) * Kt * 1024) + lane;
  const v8i* b2 = (const v8i*)(bfrag + (size_t)(nt0 + 2) * Kt * 1024) + lane;
  const v8i* b3 = (const v8i*)(bfrag + (size_t)(nt0 + 3) * Kt * 1024) + lane;
  v8i z = {0, 0, 0, 0, 0, 0, 0, 0};
  v8i c00 = z, c01 = z, c02 = z, c03 = z;
  v8i c10 = z, c11 = z, c12 = z, c13 = z;
  for (int kt = 0; kt < Kt; ++kt) {
    v8i A0 = a0[kt * 32];
    v8i A1 = a1[kt * 32];
    v8i B0 = b0[kt * 32];
    v8i B1 = b1[kt * 32];
    v8i B2 = b2[kt * 32];
    v8i B3 = b3[kt * 32];
    c00 = WMMA_IU8(A0, B0, c00);
    c01 = WMMA_IU8(A0, B1, c01);
    c02 = WMMA_IU8(A0, B2, c02);
    c03 = WMMA_IU8(A0, B3, c03);
    c10 = WMMA_IU8(A1, B0, c10);
    c11 = WMMA_IU8(A1, B1, c11);
    c12 = WMMA_IU8(A1, B2, c12);
    c13 = WMMA_IU8(A1, B3, c13);
  }
  float ws = wscale[0];
  int n4 = lane & 15;
  int hi8 = (lane >> 4) << 3;
#pragma unroll
  for (int mi = 0; mi < 2; ++mi) {
#pragma unroll
    for (int ni = 0; ni < 4; ++ni) {
      const v8i& cc = mi ? (ni == 0 ? c10 : ni == 1 ? c11 : ni == 2 ? c12 : c13)
                         : (ni == 0 ? c00 : ni == 1 ? c01 : ni == 2 ? c02 : c03);
      int n = (nt0 + ni) * 16 + n4;
#pragma unroll
      for (int r = 0; r < 8; ++r) {
        int m = (mt0 + mi) * 16 + hi8 + r;
        out[(size_t)m * D_MODEL + n] = (float)cc[r] * inv_s[m] * ws;
      }
    }
  }
}

extern "C" void kernel_launch(void* const* d_in, const int* in_sizes, int n_in,
                              void* d_out, int out_size, void* d_ws, size_t ws_size,
                              hipStream_t stream) {
  const float* x         = (const float*)d_in[0];
  const float* in_proj_w = (const float*)d_in[1];
  const float* conv_w    = (const float*)d_in[2];
  const float* conv_b    = (const float*)d_in[3];
  const float* delta_w   = (const float*)d_in[4];
  const float* delta_b   = (const float*)d_in[5];
  const float* B_w       = (const float*)d_in[6];
  const float* C_w       = (const float*)d_in[7];
  const float* out_w     = (const float*)d_in[8];
  const float* Dv        = (const float*)d_in[9];
  const float* logA      = (const float*)d_in[10];

  char* ws = (char*)d_ws;
  float* scales  = (float*)(ws + OFF_SCAL);
  float* partial = scales + 8;

  struct { const float* w; int n; int si; } reds[5] = {
    { in_proj_w, 2 * D_INNER * D_MODEL, 0 },
    { out_w,     D_MODEL * D_INNER,     1 },
    { delta_w,   D_STATE * D_INNER,     2 },
    { B_w,       D_STATE * D_INNER,     3 },
    { C_w,       D_STATE * D_INNER,     4 },
  };
  for (int i = 0; i < 5; ++i) {
    k_abs_partial<<<256, 256, 0, stream>>>(reds[i].w, reds[i].n, partial);
    k_abs_final<<<1, 256, 0, stream>>>(partial, 256, 1.f / (float)reds[i].n,
                                       scales + reds[i].si);
  }

  k_wquant<<<(2 * D_INNER * D_MODEL) / 256, 256, 0, stream>>>(
      in_proj_w, scales + 0, D_MODEL, 16, (signed char*)(ws + OFF_WIN));
  k_wquant<<<(D_MODEL * D_INNER) / 256, 256, 0, stream>>>(
      out_w, scales + 1, D_INNER, 32, (signed char*)(ws + OFF_WOUT));
  k_wquant<<<(D_STATE * D_INNER) / 256, 256, 0, stream>>>(
      delta_w, scales + 2, D_INNER, 32, (signed char*)(ws + OFF_WSML));
  k_wquant<<<(D_STATE * D_INNER) / 256, 256, 0, stream>>>(
      B_w, scales + 3, D_INNER, 32, (signed char*)(ws + OFF_WSML + 32 * 1024));
  k_wquant<<<(D_STATE * D_INNER) / 256, 256, 0, stream>>>(
      C_w, scales + 4, D_INNER, 32, (signed char*)(ws + OFF_WSML + 64 * 1024));

  k_aquant<<<MROWS, 256, 0, stream>>>(x, D_MODEL, 16,
      (signed char*)(ws + OFF_A1), (float*)(ws + OFF_IS1));
  k_gemm_gated<<<(MROWS / 32) * (D_INNER / 32) / 8, 256, 0, stream>>>(
      (const signed char*)(ws + OFF_A1), (const signed char*)(ws + OFF_WIN),
      (const float*)(ws + OFF_IS1), scales + 0, (float*)(ws + OFF_XIN));

  k_conv<<<(int)(((size_t)MROWS * D_INNER) / 256), 256, 0, stream>>>(
      (const float*)(ws + OFF_XIN), conv_w, conv_b, (float*)(ws + OFF_XCV));

  k_aquant<<<MROWS, 256, 0, stream>>>((const float*)(ws + OFF_XCV), D_INNER, 32,
      (signed char*)(ws + OFF_A2), (float*)(ws + OFF_IS2));
  k_gemm_small<<<((MROWS / 16) * 3) / 8, 256, 0, stream>>>(
      (const signed char*)(ws + OFF_A2), (const signed char*)(ws + OFF_WSML),
      (const float*)(ws + OFF_IS2), scales + 2, delta_b,
      (float*)(ws + OFF_DLT), (float*)(ws + OFF_BM), (float*)(ws + OFF_CM));

  k_scan<<<BSZ, 256, 0, stream>>>(
      (const float*)(ws + OFF_DLT), (const float*)(ws + OFF_BM),
      (const float*)(ws + OFF_CM), logA, (float*)(ws + OFF_SIG));

  k_yquant<<<MROWS, 256, 0, stream>>>(
      (const float*)(ws + OFF_XCV), (const float*)(ws + OFF_SIG), Dv,
      (signed char*)(ws + OFF_YQ), (float*)(ws + OFF_IS3));
  k_gemm_out<<<(MROWS / 32) * (D_MODEL / 64) / 8, 256, 0, stream>>>(
      (const signed char*)(ws + OFF_YQ), (const signed char*)(ws + OFF_WOUT),
      (const float*)(ws + OFF_IS3), scales + 1, (float*)d_out);

  (void)in_sizes; (void)n_in; (void)out_size; (void)ws_size;
}
